// GNNEncoder_2843268350302
// MI455X (gfx1250) — matrix operand order repaired and checked by reading, not compiled, not run on hardware
//
#include <hip/hip_runtime.h>
#include <hip/hip_bf16.h>
#include <math.h>

typedef __attribute__((ext_vector_type(16))) _Float16 v16h;
typedef __attribute__((ext_vector_type(8)))  float    v8f;

#define N_NODES 50000
#define E_EDGES 1600000
#define HDIM 64
#define LDIM 64
#define KPOOL 16

__device__ inline void wave_lds_sync() {
    __builtin_amdgcn_wave_barrier();
    asm volatile("s_wait_dscnt 0" ::: "memory");
    __builtin_amdgcn_wave_barrier();
}

__device__ inline float silu_f(float v) { return v / (1.0f + expf(-v)); }

// ---------------------------------------------------------------------------
// Fragment-major layouts (wave32, v_wmma_f32_16x16x32_f16).
// A tile (16 x 32K f16) stored as [chunks][32 lanes][16 halves]; each lane's
// fragment is one contiguous v16h (2x ds_load_b128 / global_load_b128).
__device__ inline int afrag_idx(int m, int f) {
    int chunk = f >> 5;
    int kk    = f & 31;
    int lane  = m + ((kk >> 4) << 4);
    int i     = kk & 15;
    return (chunk * 32 + lane) * 16 + i;
}

// B tile (32 x 64 f16) stored as [4 n-chunks][32 lanes][16 halves].
__device__ inline int bfrag_idx(int kp, int n) {
    int nc   = n >> 4;
    int lane = (n & 15) + ((kp >> 4) << 4);
    int i    = kp & 15;
    return (nc * 32 + lane) * 16 + i;
}

__device__ inline void zero_c(v8f c[4]) {
#pragma unroll
    for (int i = 0; i < 4; ++i) c[i] = (v8f){0.f,0.f,0.f,0.f,0.f,0.f,0.f,0.f};
}

// C(16x64) += A(16 x CHUNKS*32, LDS frag-major) @ Wp (packed f16 frag-major)
template <int CHUNKS>
__device__ inline void gemm_tile_packed(const _Float16* A, const _Float16* Wp,
                                        int lane, v8f c[4]) {
#pragma unroll
    for (int kc = 0; kc < CHUNKS; ++kc) {
        v16h a = *(const v16h*)(A + (kc * 32 + lane) * 16);
#pragma unroll
        for (int nc = 0; nc < 4; ++nc) {
            v16h b = *(const v16h*)(Wp + ((kc * 4 + nc) * 32 + lane) * 16);
            c[nc] = __builtin_amdgcn_wmma_f32_16x16x32_f16(
                false, a, false, b, (short)0, c[nc], false, false);
        }
    }
}

// Write C tile (bias + silu) into an LDS A-fragment stage (2 chunks = 16x64).
__device__ inline void store_stage_silu(_Float16* T, const v8f c[4],
                                        const float* bias, int lane) {
    int mb = (lane >> 4) << 3;
    int nb = lane & 15;
#pragma unroll
    for (int nc = 0; nc < 4; ++nc)
#pragma unroll
        for (int r = 0; r < 8; ++r) {
            int m = mb + r, n = nc * 16 + nb;
            float v = c[nc][r] + bias[n];
            T[afrag_idx(m, n)] = (_Float16)silu_f(v);
        }
}

// ---- Weight pre-pack: W (in_real x 64, f32) -> frag-major f16 --------------
__global__ void pack_weights_kernel(const float* __restrict__ W, int in_real,
                                    int kchunks, _Float16* __restrict__ out) {
    int idx = blockIdx.x * blockDim.x + threadIdx.x;
    int total = kchunks * 2048;  // kchunks * 4nc * 32lanes * 16halves
    if (idx >= total) return;
    int i = idx & 15, lane = (idx >> 4) & 31, nc = (idx >> 9) & 3, kc = idx >> 11;
    int n = nc * 16 + (lane & 15);
    int k = kc * 32 + ((lane >> 4) << 4) + i;
    out[idx] = (k < in_real) ? (_Float16)W[k * 64 + n] : (_Float16)0.0f;
}

// ---- Fused edge MLP: one wave == one 16-edge tile --------------------------
template <int HF>
__global__ __launch_bounds__(128) void edge_mlp_kernel(
    const float* __restrict__ h,   // N x HF node features
    const float* __restrict__ x,   // N x 32 (pos=[:,0:2], vel=[:,2:4])
    const int* __restrict__ src, const int* __restrict__ dst,
    const _Float16* __restrict__ ew0p, const float* __restrict__ eb0,
    const _Float16* __restrict__ ew1p, const float* __restrict__ eb1,
    const _Float16* __restrict__ ew2p, const float* __restrict__ eb2,
    const _Float16* __restrict__ vw0p, const float* __restrict__ vb0,
    const float* __restrict__ vw1, const float* __restrict__ vb1,
    float* __restrict__ m_h, float* __restrict__ m_v, int E_total) {
    constexpr int IN_REAL = 2 * HF + 2;
    constexpr int INP     = (IN_REAL + 31) & ~31;
    constexpr int CH_IN   = INP / 32;

    __shared__ alignas(32) _Float16 astage[4][CH_IN * 32 * 16];
    __shared__ alignas(32) _Float16 tstage[4][2 * 32 * 16];
    __shared__ float relstage[4][16][2];

    const int wave = threadIdx.x >> 5;
    const int lane = threadIdx.x & 31;
    const int numTiles  = (E_total + 15) >> 4;
    const int waveCount = gridDim.x * 4;

    for (int tile = blockIdx.x * 4 + wave; tile < numTiles; tile += waveCount) {
        const int ebase = tile << 4;

        int enext = ebase + (waveCount << 4);
        if (enext < E_total) {
            __builtin_prefetch(&src[enext], 0, 1);
            __builtin_prefetch(&dst[enext], 0, 1);
        }

        _Float16* Aw = &astage[wave][0];
        _Float16* T  = &tstage[wave][0];

        // ---- stage tmp = [h[dst], h[src], dist_sq, dot_vr], frag-major f16
#pragma unroll 1
        for (int j = lane; j < 16 * INP; j += 32) {
            int m = j / INP, f = j - m * INP;
            int e = ebase + m;
            float val = 0.0f;
            if (e < E_total && f < IN_REAL) {
                int s = src[e], d = dst[e];
                if (f < HF)          val = h[d * HF + f];
                else if (f < 2 * HF) val = h[s * HF + (f - HF)];
                else {
                    float rpx = x[s * 32 + 0] - x[d * 32 + 0];
                    float rpy = x[s * 32 + 1] - x[d * 32 + 1];
                    if (f == 2 * HF) {
                        val = rpx * rpx + rpy * rpy;
                    } else {
                        float rvx = x[s * 32 + 2] - x[d * 32 + 2];
                        float rvy = x[s * 32 + 3] - x[d * 32 + 3];
                        val = rvx * rpx + rvy * rpy;
                    }
                }
            }
            Aw[afrag_idx(m, f)] = (_Float16)val;
        }
        if (lane < 16) {
            int e = ebase + lane;
            float rpx = 0.f, rpy = 0.f;
            if (e < E_total) {
                int s = src[e], d = dst[e];
                rpx = x[s * 32 + 0] - x[d * 32 + 0];
                rpy = x[s * 32 + 1] - x[d * 32 + 1];
            }
            relstage[wave][lane][0] = rpx;
            relstage[wave][lane][1] = rpy;
        }
        wave_lds_sync();

        v8f c[4];

        // t = silu(tmp @ ew0 + eb0)
        zero_c(c);
        gemm_tile_packed<CH_IN>(Aw, ew0p, lane, c);
        store_stage_silu(T, c, eb0, lane);
        wave_lds_sync();

        // t = silu(t @ ew1 + eb1)
        zero_c(c);
        gemm_tile_packed<2>(T, ew1p, lane, c);
        wave_lds_sync();
        store_stage_silu(T, c, eb1, lane);
        wave_lds_sync();

        // m_h_e = t @ ew2 + eb2 ; scatter-add into m_h[dst]
        zero_c(c);
        gemm_tile_packed<2>(T, ew2p, lane, c);
        {
            int mb = (lane >> 4) << 3;
            int nb = lane & 15;
#pragma unroll
            for (int nc = 0; nc < 4; ++nc)
#pragma unroll
                for (int r = 0; r < 8; ++r) {
                    int m = mb + r, n = nc * 16 + nb;
                    int e = ebase + m;
                    if (e < E_total)
                        atomicAdd(&m_h[dst[e] * 64 + n], c[nc][r] + eb2[n]);
                }
        }

        // v = silu(tmp @ vw0 + vb0)
        zero_c(c);
        gemm_tile_packed<CH_IN>(Aw, vw0p, lane, c);
        store_stage_silu(T, c, vb0, lane);
        wave_lds_sync();

        // m_v_e = (v . vw1 + vb1) * rel_pos ; scatter-add
        if (lane < 16) {
            int e = ebase + lane;
            if (e < E_total) {
                float acc = vb1[0];
#pragma unroll 1
                for (int nn = 0; nn < 64; ++nn)
                    acc += (float)T[afrag_idx(lane, nn)] * vw1[nn];
                int d = dst[e];
                atomicAdd(&m_v[d * 2 + 0], acc * relstage[wave][lane][0]);
                atomicAdd(&m_v[d * 2 + 1], acc * relstage[wave][lane][1]);
            }
        }
        wave_lds_sync();
    }
}

// ---- Node update via WMMA: one wave == one 16-node tile --------------------
// u = silu([h, m_h, |m_v|] @ hw0 + hb0); o = short + u @ hw1 + hb1;
// h_out = LN(relu(o)).  Layer1: short = x @ sw + sb, and chunk 0 of the staged
// A tile IS x, so the shortcut GEMM reuses it, accumulating into the same C.
template <int HF, bool HAS_SW>
__global__ __launch_bounds__(128) void node_mlp_kernel(
    const float* __restrict__ h_in, const float* __restrict__ m_h,
    const float* __restrict__ m_v,
    const _Float16* __restrict__ hw0p, const float* __restrict__ hb0,
    const _Float16* __restrict__ hw1p, const float* __restrict__ hb1,
    const _Float16* __restrict__ swp,  const float* __restrict__ sb,
    const float* __restrict__ lng, const float* __restrict__ lnb,
    float* __restrict__ h_out, int n) {
    constexpr int IN_REAL = HF + 65;          // h(HF) + m_h(64) + |m_v|(1)
    constexpr int INP     = (IN_REAL + 31) & ~31;
    constexpr int CH      = INP / 32;

    __shared__ alignas(32) _Float16 astage[4][CH * 32 * 16];
    __shared__ alignas(32) _Float16 tstage[4][2 * 32 * 16];
    __shared__ float ostage[4][16][64];

    const int wave = threadIdx.x >> 5;
    const int lane = threadIdx.x & 31;
    const int numTiles  = (n + 15) >> 4;
    const int waveCount = gridDim.x * 4;

    for (int tile = blockIdx.x * 4 + wave; tile < numTiles; tile += waveCount) {
        const int base = tile << 4;
        _Float16* Aw = &astage[wave][0];
        _Float16* T  = &tstage[wave][0];
        float*    O  = &ostage[wave][0][0];

        // ---- stage concat(h, m_h, |m_v|) frag-major f16 ----
#pragma unroll 1
        for (int j = lane; j < 16 * INP; j += 32) {
            int m = j / INP, f = j - m * INP;
            int i = base + m;
            float val = 0.0f;
            if (i < n && f < IN_REAL) {
                if (f < HF)          val = h_in[i * HF + f];
                else if (f < HF + 64) val = m_h[i * 64 + (f - HF)];
                else {
                    float a = m_v[i * 2 + 0], b = m_v[i * 2 + 1];
                    val = sqrtf(a * a + b * b + 1e-12f);
                }
            }
            Aw[afrag_idx(m, f)] = (_Float16)val;
        }
        wave_lds_sync();

        v8f c[4];

        // u = silu(A @ hw0 + hb0)
        zero_c(c);
        gemm_tile_packed<CH>(Aw, hw0p, lane, c);
        store_stage_silu(T, c, hb0, lane);
        wave_lds_sync();

        // o = u @ hw1 (+ x @ sw for layer 1)
        zero_c(c);
        gemm_tile_packed<2>(T, hw1p, lane, c);
        if (HAS_SW) gemm_tile_packed<1>(Aw, swp, lane, c);  // chunk 0 of A == x

        // o += hb1 (+ sb | + h_in shortcut); relu; stage f32 for LN
        {
            int mb = (lane >> 4) << 3;
            int nb = lane & 15;
#pragma unroll
            for (int nc = 0; nc < 4; ++nc)
#pragma unroll
                for (int r = 0; r < 8; ++r) {
                    int m = mb + r, nn = nc * 16 + nb;
                    int i = base + m;
                    float v = c[nc][r] + hb1[nn];
                    if (HAS_SW) v += sb[nn];
                    else if (i < n) v += h_in[i * 64 + nn];
                    O[m * 64 + nn] = fmaxf(v, 0.f);
                }
        }
        wave_lds_sync();

        // LayerNorm per row (16 lanes, one row each)
        if (lane < 16) {
            int i = base + lane;
            if (i < n) {
                float mean = 0.f;
#pragma unroll 1
                for (int j = 0; j < 64; ++j) mean += O[lane * 64 + j];
                mean *= (1.0f / 64.0f);
                float var = 0.f;
#pragma unroll 1
                for (int j = 0; j < 64; ++j) {
                    float d = O[lane * 64 + j] - mean;
                    var += d * d;
                }
                var *= (1.0f / 64.0f);
                float inv = rsqrtf(var + 1e-5f);
#pragma unroll 1
                for (int j = 0; j < 64; ++j)
                    h_out[i * 64 + j] =
                        (O[lane * 64 + j] - mean) * inv * lng[j] + lnb[j];
            }
        }
        wave_lds_sync();
    }
}

// ---- Soft pooling via WMMA: pooled = s^T @ h accumulated in C registers ----
__global__ __launch_bounds__(256) void pool_kernel(
    const float* __restrict__ h, const float* __restrict__ x,
    const float* __restrict__ pw, const float* __restrict__ pb,
    float* __restrict__ s_out, float* __restrict__ pacc,
    float* __restrict__ macc, float* __restrict__ wacc,
    float* __restrict__ eacc, int n) {
    __shared__ alignas(32) _Float16 sstage[8][32 * 16];       // A frag (16x32)
    __shared__ alignas(32) _Float16 hstage[8][4 * 32 * 16];   // B frags (32x64)
    const int wave = threadIdx.x >> 5;
    const int lane = threadIdx.x & 31;
    const int waveTotal = gridDim.x * 8;

    v8f c[4];
    zero_c(c);
    float wpart[KPOOL], mpart[2 * KPOOL], epart = 0.f;
#pragma unroll
    for (int k = 0; k < KPOOL; ++k) { wpart[k] = 0.f; mpart[2*k] = 0.f; mpart[2*k+1] = 0.f; }

    _Float16* Sw = &sstage[wave][0];
    _Float16* Hw = &hstage[wave][0];

    const int numBatches = (n + 31) >> 5;
    for (int b = blockIdx.x * 8 + wave; b < numBatches; b += waveTotal) {
        int i = (b << 5) + lane;
        bool act = i < n;

        // fused: stage h-block to LDS B-frags while accumulating logits
        float sv[KPOOL];
#pragma unroll
        for (int k = 0; k < KPOOL; ++k) sv[k] = pb[k];
#pragma unroll 1
        for (int f = 0; f < 64; ++f) {
            float v = act ? h[i * 64 + f] : 0.f;
            Hw[bfrag_idx(lane, f)] = (_Float16)v;
#pragma unroll
            for (int k = 0; k < KPOOL; ++k) sv[k] += v * pw[f * KPOOL + k];
        }

        if (act) {
            float mx = sv[0];
#pragma unroll
            for (int k = 1; k < KPOOL; ++k) mx = fmaxf(mx, sv[k]);
            float sum = 0.f;
#pragma unroll
            for (int k = 0; k < KPOOL; ++k) { sv[k] = expf(sv[k] - mx); sum += sv[k]; }
            float rs = 1.0f / sum;
            float px = x[i * 32 + 0], py = x[i * 32 + 1];
#pragma unroll
            for (int k = 0; k < KPOOL; ++k) {
                sv[k] *= rs;
                s_out[i * KPOOL + k] = sv[k];
                wpart[k] += sv[k];
                mpart[2 * k + 0] += sv[k] * px;
                mpart[2 * k + 1] += sv[k] * py;
                epart += sv[k] * logf(sv[k] + 1e-9f);
            }
        } else {
#pragma unroll
            for (int k = 0; k < KPOOL; ++k) sv[k] = 0.f;
        }

        // stage s^T as A fragment: A[m=k][kdim=lane]
#pragma unroll
        for (int k = 0; k < KPOOL; ++k) Sw[afrag_idx(k, lane)] = (_Float16)sv[k];
        wave_lds_sync();

        v16h a = *(const v16h*)(Sw + lane * 16);
#pragma unroll
        for (int nc = 0; nc < 4; ++nc) {
            v16h bb = *(const v16h*)(Hw + (nc * 32 + lane) * 16);
            c[nc] = __builtin_amdgcn_wmma_f32_16x16x32_f16(
                false, a, false, bb, (short)0, c[nc], false, false);
        }
        wave_lds_sync();
    }

    // flush partial pooled matrix + scalar partials
    {
        int mb = (lane >> 4) << 3;
        int nb = lane & 15;
#pragma unroll
        for (int nc = 0; nc < 4; ++nc)
#pragma unroll
            for (int r = 0; r < 8; ++r)
                atomicAdd(&pacc[(mb + r) * 64 + nc * 16 + nb], c[nc][r]);
    }
#pragma unroll
    for (int k = 0; k < KPOOL; ++k) {
        atomicAdd(&wacc[k], wpart[k]);
        atomicAdd(&macc[2 * k + 0], mpart[2 * k + 0]);
        atomicAdd(&macc[2 * k + 1], mpart[2 * k + 1]);
    }
    atomicAdd(eacc, epart);
}

// ---- Final: pooled/mu/ent + output MLP ------------------------------------
__global__ __launch_bounds__(256) void final_kernel(
    const float* __restrict__ pacc, const float* __restrict__ macc,
    const float* __restrict__ wacc, const float* __restrict__ eacc,
    const float* __restrict__ w0, const float* __restrict__ b0,
    const float* __restrict__ w1, const float* __restrict__ b1,
    const float* __restrict__ gain,
    float* __restrict__ latent, float* __restrict__ mu,
    float* __restrict__ ent, int n) {
    __shared__ float pooled[KPOOL * 64];
    __shared__ float t1[KPOOL * 64];
    int tid = threadIdx.x;
    for (int idx = tid; idx < KPOOL * 64; idx += 256) {
        int k = idx >> 6;
        pooled[idx] = pacc[idx] / (wacc[k] + 1e-8f);
    }
    if (tid < KPOOL * 2) mu[tid] = macc[tid] / (wacc[tid >> 1] + 1e-8f);
    if (tid == 0) ent[0] = -eacc[0] / (float)n;
    __syncthreads();
    for (int idx = tid; idx < KPOOL * 64; idx += 256) {
        int k = idx >> 6, j = idx & 63;
        float a = b0[j];
#pragma unroll 1
        for (int f = 0; f < 64; ++f) a += pooled[k * 64 + f] * w0[f * 64 + j];
        t1[idx] = fmaxf(a, 0.f);
    }
    __syncthreads();
    for (int idx = tid; idx < KPOOL * 64; idx += 256) {
        int k = idx >> 6, l = idx & 63;
        float a = b1[l];
#pragma unroll 1
        for (int j = 0; j < 64; ++j) a += t1[k * 64 + j] * w1[j * 64 + l];
        latent[idx] = a * gain[l];
    }
}

__global__ void zero_kernel(float* __restrict__ p, int n) {
    int i = blockIdx.x * blockDim.x + threadIdx.x;
    if (i < n) p[i] = 0.f;
}

extern "C" void kernel_launch(void* const* d_in, const int* in_sizes, int n_in,
                              void* d_out, int out_size, void* d_ws, size_t ws_size,
                              hipStream_t stream) {
    const int N = N_NODES, E = E_EDGES;

    const float* x   = (const float*)d_in[0];
    const int*   ei  = (const int*)d_in[1];
    const int*   src = ei;
    const int*   dst = ei + E;
    const float* g1_ew0 = (const float*)d_in[3];
    const float* g1_eb0 = (const float*)d_in[4];
    const float* g1_ew1 = (const float*)d_in[5];
    const float* g1_eb1 = (const float*)d_in[6];
    const float* g1_ew2 = (const float*)d_in[7];
    const float* g1_eb2 = (const float*)d_in[8];
    const float* g1_vw0 = (const float*)d_in[9];
    const float* g1_vb0 = (const float*)d_in[10];
    const float* g1_vw1 = (const float*)d_in[11];
    const float* g1_vb1 = (const float*)d_in[12];
    const float* g1_hw0 = (const float*)d_in[13];
    const float* g1_hb0 = (const float*)d_in[14];
    const float* g1_hw1 = (const float*)d_in[15];
    const float* g1_hb1 = (const float*)d_in[16];
    const float* g1_sw  = (const float*)d_in[17];
    const float* g1_sb  = (const float*)d_in[18];
    const float* ln1_g  = (const float*)d_in[19];
    const float* ln1_b  = (const float*)d_in[20];
    const float* g2_ew0 = (const float*)d_in[21];
    const float* g2_eb0 = (const float*)d_in[22];
    const float* g2_ew1 = (const float*)d_in[23];
    const float* g2_eb1 = (const float*)d_in[24];
    const float* g2_ew2 = (const float*)d_in[25];
    const float* g2_eb2 = (const float*)d_in[26];
    const float* g2_vw0 = (const float*)d_in[27];
    const float* g2_vb0 = (const float*)d_in[28];
    const float* g2_vw1 = (const float*)d_in[29];
    const float* g2_vb1 = (const float*)d_in[30];
    const float* g2_hw0 = (const float*)d_in[31];
    const float* g2_hb0 = (const float*)d_in[32];
    const float* g2_hw1 = (const float*)d_in[33];
    const float* g2_hb1 = (const float*)d_in[34];
    const float* ln2_g  = (const float*)d_in[35];
    const float* ln2_b  = (const float*)d_in[36];
    const float* pool_w = (const float*)d_in[37];
    const float* pool_b = (const float*)d_in[38];
    const float* out_w0 = (const float*)d_in[39];
    const float* out_b0 = (const float*)d_in[40];
    const float* out_w1 = (const float*)d_in[41];
    const float* out_b1 = (const float*)d_in[42];
    const float* gain   = (const float*)d_in[43];

    // ---- workspace layout (floats) ----
    float* ws   = (float*)d_ws;
    float* m_h  = ws;                    // N*64
    float* m_v  = m_h + (size_t)N * 64;  // N*2 (contiguous with m_h)
    float* h1   = m_v + (size_t)N * 2;   // N*64
    float* h2   = h1 + (size_t)N * 64;   // N*64
    float* pacc = h2 + (size_t)N * 64;   // 16*64
    float* macc = pacc + KPOOL * 64;     // 16*2
    float* wacc = macc + KPOOL * 2;      // 16
    float* eacc = wacc + KPOOL;          // 1

    size_t packOffF = ((size_t)N * 194 + 1073 + 7) & ~(size_t)7;  // 32B aligned
    _Float16* wpack = (_Float16*)(ws + packOffF);
    // packed sizes (halves): chunks*2048
    _Float16* l1_ew0p = wpack;                 // 3 chunks -> 6144
    _Float16* l1_ew1p = l1_ew0p + 6144;        // 2 -> 4096
    _Float16* l1_ew2p = l1_ew1p + 4096;        // 2 -> 4096
    _Float16* l1_vw0p = l1_ew2p + 4096;        // 3 -> 6144
    _Float16* l2_ew0p = l1_vw0p + 6144;        // 5 -> 10240
    _Float16* l2_ew1p = l2_ew0p + 10240;       // 2 -> 4096
    _Float16* l2_ew2p = l2_ew1p + 4096;        // 2 -> 4096
    _Float16* l2_vw0p = l2_ew2p + 4096;        // 5 -> 10240
    _Float16* l1_hw0p = l2_vw0p + 10240;       // 97 -> 4 chunks -> 8192
    _Float16* l1_hw1p = l1_hw0p + 8192;        // 2 -> 4096
    _Float16* l1_swp  = l1_hw1p + 4096;        // 32 -> 1 chunk -> 2048
    _Float16* l2_hw0p = l1_swp + 2048;         // 129 -> 5 chunks -> 10240
    _Float16* l2_hw1p = l2_hw0p + 10240;       // 2 -> 4096

    // ---- output layout: latent[16*64], s[N*16], ent[1], mu[16*2] ----
    float* out    = (float*)d_out;
    float* latent = out;
    float* s_out  = out + KPOOL * LDIM;
    float* entp   = s_out + (size_t)N * KPOOL;
    float* mu     = entp + 1;

    // ---- pack all weight matrices to fragment-major f16 (once per launch) --
    pack_weights_kernel<<<(3 * 2048 + 255) / 256, 256, 0, stream>>>(g1_ew0, 66, 3, l1_ew0p);
    pack_weights_kernel<<<(2 * 2048 + 255) / 256, 256, 0, stream>>>(g1_ew1, 64, 2, l1_ew1p);
    pack_weights_kernel<<<(2 * 2048 + 255) / 256, 256, 0, stream>>>(g1_ew2, 64, 2, l1_ew2p);
    pack_weights_kernel<<<(3 * 2048 + 255) / 256, 256, 0, stream>>>(g1_vw0, 66, 3, l1_vw0p);
    pack_weights_kernel<<<(5 * 2048 + 255) / 256, 256, 0, stream>>>(g2_ew0, 130, 5, l2_ew0p);
    pack_weights_kernel<<<(2 * 2048 + 255) / 256, 256, 0, stream>>>(g2_ew1, 64, 2, l2_ew1p);
    pack_weights_kernel<<<(2 * 2048 + 255) / 256, 256, 0, stream>>>(g2_ew2, 64, 2, l2_ew2p);
    pack_weights_kernel<<<(5 * 2048 + 255) / 256, 256, 0, stream>>>(g2_vw0, 130, 5, l2_vw0p);
    pack_weights_kernel<<<(4 * 2048 + 255) / 256, 256, 0, stream>>>(g1_hw0, 97, 4, l1_hw0p);
    pack_weights_kernel<<<(2 * 2048 + 255) / 256, 256, 0, stream>>>(g1_hw1, 64, 2, l1_hw1p);
    pack_weights_kernel<<<(1 * 2048 + 255) / 256, 256, 0, stream>>>(g1_sw, 32, 1, l1_swp);
    pack_weights_kernel<<<(5 * 2048 + 255) / 256, 256, 0, stream>>>(g2_hw0, 129, 5, l2_hw0p);
    pack_weights_kernel<<<(2 * 2048 + 255) / 256, 256, 0, stream>>>(g2_hw1, 64, 2, l2_hw1p);

    const int EDGE_BLOCKS = 4096;
    const int NODE_BLOCKS = 512;

    // ---------- Layer 1 ----------
    zero_kernel<<<(N * 66 + 255) / 256, 256, 0, stream>>>(m_h, N * 66);
    edge_mlp_kernel<32><<<EDGE_BLOCKS, 128, 0, stream>>>(
        x, x, src, dst,
        l1_ew0p, g1_eb0, l1_ew1p, g1_eb1, l1_ew2p, g1_eb2,
        l1_vw0p, g1_vb0, g1_vw1, g1_vb1, m_h, m_v, E);
    node_mlp_kernel<32, true><<<NODE_BLOCKS, 128, 0, stream>>>(
        x, m_h, m_v, l1_hw0p, g1_hb0, l1_hw1p, g1_hb1,
        l1_swp, g1_sb, ln1_g, ln1_b, h1, N);

    // ---------- Layer 2 ----------
    zero_kernel<<<(N * 66 + 255) / 256, 256, 0, stream>>>(m_h, N * 66);
    edge_mlp_kernel<64><<<EDGE_BLOCKS, 128, 0, stream>>>(
        h1, x, src, dst,
        l2_ew0p, g2_eb0, l2_ew1p, g2_eb1, l2_ew2p, g2_eb2,
        l2_vw0p, g2_vb0, g2_vw1, g2_vb1, m_h, m_v, E);
    node_mlp_kernel<64, false><<<NODE_BLOCKS, 128, 0, stream>>>(
        h1, m_h, m_v, l2_hw0p, g2_hb0, l2_hw1p, g2_hb1,
        nullptr, nullptr, ln2_g, ln2_b, h2, N);

    // ---------- Pool + head ----------
    zero_kernel<<<(KPOOL * 64 + KPOOL * 2 + KPOOL + 1 + 255) / 256, 256, 0, stream>>>(
        pacc, KPOOL * 64 + KPOOL * 2 + KPOOL + 1);
    pool_kernel<<<64, 256, 0, stream>>>(h2, x, pool_w, pool_b,
                                        s_out, pacc, macc, wacc, eacc, N);
    final_kernel<<<1, 256, 0, stream>>>(pacc, macc, wacc, eacc,
                                        out_w0, out_b0, out_w1, out_b1, gain,
                                        latent, mu, entp, N);
}
